// SpectralEncoder_52510270161140
// MI455X (gfx1250) — matrix operand order repaired and checked by reading, not compiled, not run on hardware
//
#include <hip/hip_runtime.h>
#include <hip/hip_bf16.h>
#include <math.h>

typedef __attribute__((ext_vector_type(16))) _Float16     v16h;
typedef __attribute__((ext_vector_type(8)))  float        v8f;
typedef __attribute__((ext_vector_type(4)))  unsigned int v4u;

union ABv { v16h h; v4u u[2]; };

#define H      512
#define TB     16      // batch items per workgroup
#define NTILES 9       // 3 entity M-tiles + 6 pad M-tiles (16 rows each)

// ---- dynamic LDS layout (bytes) ----
#define TOK_OFF    0                               // float  [16][10][10]   6400 B
#define H1_OFF     6400                            // f16    [9][16][512] 147456 B
#define H2_OFF     (H1_OFF + NTILES*TB*H*2)        // f16    [9][16][512] 147456 B
#define AGG_OFF    (H2_OFF + NTILES*TB*H*2)        // f16    [16][512]     16384 B
#define POS_OFF    (AGG_OFF + TB*H*2)              // float  [16][4][3]      768 B
#define COEF_OFF   (POS_OFF + TB*4*3*4)            // float  [16][4][9]     2304 B
#define COV_OFF    (COEF_OFF + TB*4*9*4)           // float  [16][4][9]     2304 B
#define LOGIT_OFF  (COV_OFF + TB*4*9*4)            // float  [16][6]         384 B
#define SMEM_BYTES (LOGIT_OFF + TB*6*4)            // = 323456 B  (< 320 KiB WGP LDS)

__device__ __forceinline__ float softplusf(float x) {
    return (x > 20.f) ? x : log1pf(expf(x));
}

// ---------------------------------------------------------------------------
// Prep: transpose + f16-convert the two big weight matrices.
//   WtE[e][n][k] = (f16) ent_W2[e][k][n]      (3 x 512 x 512)
//   WtP[n][k]    = (f16) pad_W2[k][n]         (512 x 512)
// ---------------------------------------------------------------------------
__global__ void se_prep(const float* __restrict__ entW2,
                        const float* __restrict__ padW2,
                        _Float16* __restrict__ WtE,
                        _Float16* __restrict__ WtP) {
    size_t i = (size_t)blockIdx.x * 256 + threadIdx.x;
    const size_t NE = (size_t)3 * H * H;
    if (i < NE) {
        int e   = (int)(i / (H * H));
        int rem = (int)(i % (H * H));
        int n   = rem / H;
        int k   = rem % H;
        WtE[i] = (_Float16)entW2[((size_t)e * H + k) * H + n];
    } else {
        size_t j = i - NE;
        if (j < (size_t)H * H) {
            int n = (int)(j / H), k = (int)(j % H);
            WtP[j] = (_Float16)padW2[(size_t)k * H + n];
        }
    }
}

// ---------------------------------------------------------------------------
// Main fused kernel: one workgroup = 16 batch items.
// ---------------------------------------------------------------------------
__global__ __launch_bounds__(256)
void se_main(const float* __restrict__ tokens,
             const float* __restrict__ ent_W1,  const float* __restrict__ ent_b1,
             const float* __restrict__ ent_b2,
             const float* __restrict__ ent_Wpos,const float* __restrict__ ent_bpos,
             const float* __restrict__ ent_Wc,  const float* __restrict__ ent_bc,
             const float* __restrict__ ent_Wv,  const float* __restrict__ ent_bv,
             const float* __restrict__ pad_W1,  const float* __restrict__ pad_b1,
             const float* __restrict__ pad_b2,
             const float* __restrict__ attn_W,  const float* __restrict__ attn_b,
             const float* __restrict__ b_Wpos,  const float* __restrict__ b_bpos,
             const float* __restrict__ b_Wc,    const float* __restrict__ b_bc,
             const float* __restrict__ b_Wv,    const float* __restrict__ b_bv,
             const float* __restrict__ gs_W,    const float* __restrict__ gs_b,
             const _Float16* __restrict__ WtE,  const _Float16* __restrict__ WtP,
             float* __restrict__ out) {
    extern __shared__ char smem[];
    float*    tok   = (float*)   (smem + TOK_OFF);
    _Float16* h1    = (_Float16*)(smem + H1_OFF);
    _Float16* h2    = (_Float16*)(smem + H2_OFF);
    _Float16* agg   = (_Float16*)(smem + AGG_OFF);
    float*    posA  = (float*)   (smem + POS_OFF);
    float*    coefA = (float*)   (smem + COEF_OFF);
    float*    covA  = (float*)   (smem + COV_OFF);
    float*    logit = (float*)   (smem + LOGIT_OFF);

    const int tid  = threadIdx.x;
    const int wave = tid >> 5;
    const int lane = tid & 31;
    const int b0   = blockIdx.x * TB;

    // ---- stage tokens (16 x 100 f32, coalesced) ----
    for (int i = tid; i < TB * 100; i += 256)
        tok[i] = tokens[(size_t)b0 * 100 + i];
    __syncthreads();

    // ---- game head (token row 9): out[.., 102..104] ----
    if (tid < TB * 3) {
        int b = tid / 3, p = tid % 3;
        float s = gs_b[p];
        #pragma unroll
        for (int t = 0; t < 10; ++t) s = fmaf(tok[b * 100 + 90 + t], gs_W[t * 3 + p], s);
        out[(size_t)(b0 + b) * 105 + 102 + p] = s;
    }

    // ---- layer 1 (K=10) f32 VALU -> relu -> f16 into h1 LDS ----
    // row r = tile*16 + b ; tiles 0..2 = entity e (token row e, weight ent_W1[e])
    //                        tiles 3..8 = pad j   (token row 3+j, weight pad_W1)
    for (int idx = tid; idx < NTILES * TB * H; idx += 256) {
        int k = idx & (H - 1);
        int r = idx >> 9;
        int t = r >> 4;
        int b = r & 15;
        const float* W1; const float* bias;
        if (t < 3) { W1 = ent_W1 + t * 10 * H; bias = ent_b1 + t * H; }
        else       { W1 = pad_W1;              bias = pad_b1; }
        const float* x = tok + b * 100 + t * 10;   // token row index == tile index
        float s = bias[k];
        #pragma unroll
        for (int j = 0; j < 10; ++j) s = fmaf(x[j], W1[j * H + k], s);
        h1[idx] = (_Float16)fmaxf(s, 0.f);
    }
    __syncthreads();

    // ---- big GEMM: h2 = relu(h1 @ W2 + b2) via v_wmma_f32_16x16x32_f16 ----
    // Each wave owns 4 N-tiles; B tile (per weight matrix) reused across 9 M-tiles.
    const int nlo = lane & 15;   // N index within tile (B/C/D lane = column)
    const int hi  = lane >> 4;   // K-half / M-half selector
    for (int ntv = 0; ntv < 4; ++ntv) {
        const int nt = wave * 4 + ntv;
        const int n  = nt * 16 + nlo;
        v8f acc[NTILES] = {};
        for (int kk = 0; kk < H / 32; ++kk) {
            const int kbase = kk * 32;
            // B operands: lane = col n, 16 consecutive k (contiguous in Wt[n][k])
            ABv Bt[4];
            const int koff = kbase + hi * 16;
            #pragma unroll
            for (int e = 0; e < 3; ++e) {
                const _Float16* p = WtE + (((size_t)e * H + n) * H + koff);
                Bt[e].u[0] = *(const v4u*)(p);
                Bt[e].u[1] = *(const v4u*)(p + 8);
            }
            {
                const _Float16* p = WtP + ((size_t)n * H + koff);
                Bt[3].u[0] = *(const v4u*)(p);
                Bt[3].u[1] = *(const v4u*)(p + 8);
            }
            #pragma unroll
            for (int mt = 0; mt < NTILES; ++mt) {
                // A operand: lane = row (batch), 16 contiguous K per lane half, same order as B
                const _Float16* ap = h1 + ((size_t)(mt * 16 + nlo) * H + kbase + hi * 16);
                ABv At;
                At.u[0] = *(const v4u*)(ap);        // K = kbase + hi*16 .. +7
                At.u[1] = *(const v4u*)(ap + 8);    // K = kbase + hi*16 + 8 .. +15
                const v16h Bv = (mt < 3) ? Bt[mt].h : Bt[3].h;
                acc[mt] = __builtin_amdgcn_wmma_f32_16x16x32_f16(
                    false, At.h, false, Bv, (short)0, acc[mt], false, false);
            }
        }
        // epilogue: +bias, relu, f16 -> h2 LDS
        #pragma unroll
        for (int mt = 0; mt < NTILES; ++mt) {
            const float bias = (mt < 3) ? ent_b2[mt * H + n] : pad_b2[n];
            #pragma unroll
            for (int v = 0; v < 8; ++v) {
                const int m = hi * 8 + v;           // C/D layout: lanes16-31 hold M=8..15
                float val = fmaxf(acc[mt][v] + bias, 0.f);
                h2[(size_t)(mt * 16 + m) * H + n] = (_Float16)val;
            }
        }
    }
    __syncthreads();

    // ---- entity heads: 48 (b,e) pairs x 21 dots of length 512 ----
    for (int idx = tid; idx < 48 * 21; idx += 256) {
        int d = idx % 21, be = idx / 21;
        int b = be / 3, e = be % 3;
        const _Float16* hr = h2 + (size_t)(e * 16 + b) * H;
        const float* W; int stride; float s;
        if (d < 3)       { W = ent_Wpos + (size_t)e*H*3 + d;      stride = 3; s = ent_bpos[e*3+d]; }
        else if (d < 12) { W = ent_Wc   + (size_t)e*H*9 + (d-3);  stride = 9; s = ent_bc[e*9+d-3]; }
        else             { W = ent_Wv   + (size_t)e*H*9 + (d-12); stride = 9; s = ent_bv[e*9+d-12]; }
        #pragma unroll 8
        for (int h = 0; h < H; ++h) s = fmaf((float)hr[h], W[(size_t)h * stride], s);
        if (d < 3)       posA [(b*4 + e)*3 + d]        = s;
        else if (d < 12) coefA[(b*4 + e)*9 + (d - 3)]  = s;
        else             covA [(b*4 + e)*9 + (d - 12)] = softplusf(s);
    }

    // ---- attention logits over the 6 pad rows ----
    for (int idx = tid; idx < TB * 6; idx += 256) {
        int b = idx / 6, j = idx % 6;
        const _Float16* hr = h2 + (size_t)((3 + j) * 16 + b) * H;
        float s = attn_b[0];
        #pragma unroll 8
        for (int h = 0; h < H; ++h) s = fmaf((float)hr[h], attn_W[h], s);
        logit[b * 6 + j] = s;
    }
    __syncthreads();

    // ---- softmax over pad rows (one thread per batch item) ----
    if (tid < TB) {
        float mx = -1e30f;
        #pragma unroll
        for (int j = 0; j < 6; ++j) mx = fmaxf(mx, logit[tid * 6 + j]);
        float w[6], sum = 0.f;
        #pragma unroll
        for (int j = 0; j < 6; ++j) { w[j] = expf(logit[tid * 6 + j] - mx); sum += w[j]; }
        const float inv = 1.f / sum;
        #pragma unroll
        for (int j = 0; j < 6; ++j) logit[tid * 6 + j] = w[j] * inv;
    }
    __syncthreads();

    // ---- agg = sum_j wts * hp ----
    for (int idx = tid; idx < TB * H; idx += 256) {
        int b = idx >> 9, h = idx & (H - 1);
        float s = 0.f;
        #pragma unroll
        for (int j = 0; j < 6; ++j)
            s = fmaf(logit[b * 6 + j], (float)h2[(size_t)((3 + j) * 16 + b) * H + h], s);
        agg[(size_t)b * H + h] = (_Float16)s;
    }
    __syncthreads();

    // ---- pad-branch heads: 16 x 21 dots ----
    for (int idx = tid; idx < TB * 21; idx += 256) {
        int b = idx / 21, d = idx % 21;
        const _Float16* ar = agg + (size_t)b * H;
        const float* W; int stride; float s;
        if (d < 3)       { W = b_Wpos + d;      stride = 3; s = b_bpos[d]; }
        else if (d < 12) { W = b_Wc + (d - 3);  stride = 9; s = b_bc[d - 3]; }
        else             { W = b_Wv + (d - 12); stride = 9; s = b_bv[d - 12]; }
        #pragma unroll 8
        for (int h = 0; h < H; ++h) s = fmaf((float)ar[h], W[(size_t)h * stride], s);
        if (d < 3)       posA [(b*4 + 3)*3 + d]        = s;
        else if (d < 12) coefA[(b*4 + 3)*9 + (d - 3)]  = s;
        else             covA [(b*4 + 3)*9 + (d - 12)] = softplusf(s);
    }
    __syncthreads();

    // ---- geometric epilogue + output assembly (one thread per batch item) ----
    if (tid < TB) {
        const int b = tid;
        const size_t ob = (size_t)(b0 + b) * 105;
        float px[4], py[4], pz[4];
        #pragma unroll
        for (int i = 0; i < 4; ++i) {
            px[i] = posA[(b*4 + i)*3 + 0];
            py[i] = posA[(b*4 + i)*3 + 1];
            pz[i] = posA[(b*4 + i)*3 + 2];
        }
        #pragma unroll
        for (int i = 0; i < 4; ++i) {
            float* o = out + ob + i * 24;
            #pragma unroll
            for (int c = 0; c < 9; ++c) o[c]     = coefA[(b*4 + i)*9 + c];
            #pragma unroll
            for (int c = 0; c < 9; ++c) o[9 + c] = covA [(b*4 + i)*9 + c];
            o[18] = (4096.f - px[i]) * 1e-3f;
            o[19] = (4096.f + px[i]) * 1e-3f;
            o[20] = (5120.f - py[i]) * 1e-3f;
            o[21] = (5120.f + py[i]) * 1e-3f;
            o[22] = (pz[i] - 0.f)    * 1e-3f;
            o[23] = (2044.f - pz[i]) * 1e-3f;
        }
        const int iu[6] = {0,0,0,1,1,2};
        const int ju[6] = {1,2,3,2,3,3};
        #pragma unroll
        for (int t = 0; t < 6; ++t) {
            const int i = iu[t], j = ju[t];
            const float dx = px[i]-px[j], dy = py[i]-py[j], dz = pz[i]-pz[j];
            const float d2 = dx*dx + dy*dy + dz*dz;
            const float aff = expf(-2.f * d2);      // exp(-d2 / (2*0.5^2))
            float cd = 0.f;
            #pragma unroll
            for (int c = 0; c < 9; ++c)
                cd = fmaf(coefA[(b*4 + i)*9 + c], coefA[(b*4 + j)*9 + c], cd);
            out[ob + 96 + t] = aff * cd;            // TAU == 1
        }
    }
}

// ---------------------------------------------------------------------------
extern "C" void kernel_launch(void* const* d_in, const int* in_sizes, int n_in,
                              void* d_out, int out_size, void* d_ws, size_t ws_size,
                              hipStream_t stream) {
    const float* tokens   = (const float*)d_in[0];
    const float* ent_W1   = (const float*)d_in[1];
    const float* ent_b1   = (const float*)d_in[2];
    const float* ent_W2   = (const float*)d_in[3];
    const float* ent_b2   = (const float*)d_in[4];
    const float* ent_Wpos = (const float*)d_in[5];
    const float* ent_bpos = (const float*)d_in[6];
    const float* ent_Wc   = (const float*)d_in[7];
    const float* ent_bc   = (const float*)d_in[8];
    const float* ent_Wv   = (const float*)d_in[9];
    const float* ent_bv   = (const float*)d_in[10];
    const float* pad_W1   = (const float*)d_in[11];
    const float* pad_b1   = (const float*)d_in[12];
    const float* pad_W2   = (const float*)d_in[13];
    const float* pad_b2   = (const float*)d_in[14];
    const float* attn_W   = (const float*)d_in[15];
    const float* attn_b   = (const float*)d_in[16];
    const float* b_Wpos   = (const float*)d_in[17];
    const float* b_bpos   = (const float*)d_in[18];
    const float* b_Wc     = (const float*)d_in[19];
    const float* b_bc     = (const float*)d_in[20];
    const float* b_Wv     = (const float*)d_in[21];
    const float* b_bv     = (const float*)d_in[22];
    const float* gs_W     = (const float*)d_in[23];
    const float* gs_b     = (const float*)d_in[24];

    _Float16* WtE = (_Float16*)d_ws;                                       // 3*512*512 f16
    _Float16* WtP = (_Float16*)((char*)d_ws + (size_t)3 * H * H * 2);      // 512*512 f16

    const int B = in_sizes[0] / 100;          // tokens is (B,10,10)

    se_prep<<<(4 * H * H) / 256, 256, 0, stream>>>(ent_W2, pad_W2, WtE, WtP);

    se_main<<<B / TB, 256, SMEM_BYTES, stream>>>(
        tokens,
        ent_W1, ent_b1, ent_b2,
        ent_Wpos, ent_bpos, ent_Wc, ent_bc, ent_Wv, ent_bv,
        pad_W1, pad_b1, pad_b2,
        attn_W, attn_b,
        b_Wpos, b_bpos, b_Wc, b_bc, b_Wv, b_bv,
        gs_W, gs_b,
        WtE, WtP,
        (float*)d_out);
}
